// MultiHeadAttention_50629074485891
// MI455X (gfx1250) — matrix operand order repaired and checked by reading, not compiled, not run on hardware
//
#include <hip/hip_runtime.h>
#include <hip/hip_bf16.h>

#define B_   2
#define T_   2048
#define DIN  2048
#define DOUT 2048
#define H_   16
#define KVH_ 4
#define HD_  128

typedef __attribute__((ext_vector_type(16))) _Float16 v16h;
typedef __attribute__((ext_vector_type(8)))  _Float16 v8h;
typedef __attribute__((ext_vector_type(4)))  _Float16 v4h;
typedef __attribute__((ext_vector_type(8)))  float    v8f;

static __device__ __forceinline__ v16h cat8(v8h lo, v8h hi) {
  return __builtin_shufflevector(lo, hi, 0,1,2,3,4,5,6,7,8,9,10,11,12,13,14,15);
}
static __device__ __forceinline__ v16h load16(const _Float16* p) {
  v8h lo = *(const v8h*)p;
  v8h hi = *(const v8h*)(p + 8);
  return cat8(lo, hi);
}
static __device__ __forceinline__ v8f wmma16(v16h a, v16h b, v8f c) {
  // D = A(16x32 f16) * B(32x16 f16) + C(16x16 f32)
  return __builtin_amdgcn_wmma_f32_16x16x32_f16(false, a, false, b, (short)0, c,
                                                false, false);
}

// Async global->LDS DMA (ASYNCcnt-tracked), 16B per lane per instruction.
// ldsoff is the workgroup-relative LDS byte offset (low 32 bits of the
// generic pointer for shared memory on AMDGPU).
static __device__ __forceinline__ void async_ld16(const _Float16* gsrc,
                                                  unsigned ldsoff) {
  asm volatile("global_load_async_to_lds_b128 %0, %1, off"
               :: "v"(ldsoff),
                  "v"((unsigned long long)(uintptr_t)gsrc)
               : "memory");
}
static __device__ __forceinline__ void wait_async(int pending) {
  if (pending) asm volatile("s_wait_asynccnt 0x4" ::: "memory");
  else         asm volatile("s_wait_asynccnt 0x0" ::: "memory");
}

// ---------------------------------------------------------------------------
// Generic WMMA GEMM: C[MxN] = A[MxK] @ B[KxN]  (A f32 or f16; C f16 or f32+bias)
// Block tile 128x128, K-tile 32, 256 threads = 8 waves in a 2x4 wave grid.
// A staged row-major f16 in LDS (stride 48 halves = 96B, 16B-aligned chunks).
// B staged TRANSPOSED (N x K) so B-fragments are contiguous 32B per lane.
// (f32->f16 conversion happens through VGPRs, so plain loads, not async DMA.)
// ---------------------------------------------------------------------------
template<bool A_HALF, bool OUT_F32>
__global__ __launch_bounds__(256)
void gemm_wmma(const void* __restrict__ Aptr, const float* __restrict__ Bmat,
               void* __restrict__ Cout, const float* __restrict__ bias,
               int M, int N, int K)
{
  __shared__ __align__(16) _Float16 As[128 * 48];
  __shared__ __align__(16) _Float16 Bs[128 * 48];
  const int t    = threadIdx.x;
  const int bm   = blockIdx.y * 128;
  const int bn   = blockIdx.x * 128;
  const int wid  = t >> 5, lane = t & 31;
  const int wm   = (wid >> 2) * 64;   // 2 waves along M, 4 tiles each
  const int wn   = (wid & 3) * 32;    // 4 waves along N, 2 tiles each
  const int lrow = lane & 15, sel = lane >> 4;

  const v8f zf = {0.f,0.f,0.f,0.f,0.f,0.f,0.f,0.f};
  v8f acc[4][2];
  #pragma unroll
  for (int mi = 0; mi < 4; ++mi)
    #pragma unroll
    for (int ni = 0; ni < 2; ++ni) acc[mi][ni] = zf;

  for (int k0 = 0; k0 < K; k0 += 32) {
    // ---- stage A tile (128 x 32) ----
    if constexpr (!A_HALF) {
      const float* A = (const float*)Aptr;
      #pragma unroll
      for (int c = 0; c < 4; ++c) {
        int idx = t + c * 256;            // 1024 float4 chunks
        int r   = idx >> 3;
        int c4  = (idx & 7) << 2;
        float4 f = *(const float4*)(A + (size_t)(bm + r) * K + k0 + c4);
        v4h h4 = { (_Float16)f.x, (_Float16)f.y, (_Float16)f.z, (_Float16)f.w };
        *(v4h*)(&As[r * 48 + c4]) = h4;
        if (k0 + 32 < K)
          __builtin_prefetch(A + (size_t)(bm + r) * K + k0 + 32 + c4, 0, 1);
      }
    } else {
      const _Float16* A = (const _Float16*)Aptr;
      #pragma unroll
      for (int c = 0; c < 2; ++c) {
        int idx = t + c * 256;            // 512 chunks of 8 halves
        int r   = idx >> 2;
        int c8  = (idx & 3) << 3;
        *(v8h*)(&As[r * 48 + c8]) =
            *(const v8h*)(A + (size_t)(bm + r) * K + k0 + c8);
        if (k0 + 32 < K)
          __builtin_prefetch(A + (size_t)(bm + r) * K + k0 + 32 + c8, 0, 1);
      }
    }
    // ---- stage B tile (32 x 128), transposed into Bs[N][K] ----
    #pragma unroll
    for (int c = 0; c < 4; ++c) {
      int idx = t + c * 256;              // 1024 float4 chunks
      int kr  = idx >> 5;
      int n4  = (idx & 31) << 2;
      float4 f = *(const float4*)(Bmat + (size_t)(k0 + kr) * N + bn + n4);
      Bs[(n4 + 0) * 48 + kr] = (_Float16)f.x;
      Bs[(n4 + 1) * 48 + kr] = (_Float16)f.y;
      Bs[(n4 + 2) * 48 + kr] = (_Float16)f.z;
      Bs[(n4 + 3) * 48 + kr] = (_Float16)f.w;
      if (k0 + 32 < K)
        __builtin_prefetch(Bmat + (size_t)(k0 + 32 + kr) * N + bn + n4, 0, 1);
    }
    __syncthreads();

    // ---- fragments + 8 WMMA ----
    v16h a[4], b[2];
    #pragma unroll
    for (int mi = 0; mi < 4; ++mi) {
      const _Float16* base = &As[(wm + mi * 16 + lrow) * 48];
      a[mi] = cat8(*(const v8h*)(base + sel * 16),
                   *(const v8h*)(base + sel * 16 + 8));
    }
    #pragma unroll
    for (int ni = 0; ni < 2; ++ni) {
      const _Float16* base = &Bs[(wn + ni * 16 + lrow) * 48 + sel * 16];
      b[ni] = cat8(*(const v8h*)(base), *(const v8h*)(base + 8));
    }
    #pragma unroll
    for (int mi = 0; mi < 4; ++mi)
      #pragma unroll
      for (int ni = 0; ni < 2; ++ni)
        acc[mi][ni] = wmma16(a[mi], b[ni], acc[mi][ni]);
    __syncthreads();
  }

  // ---- epilogue: C layout VGPR i -> row (i or i+8 per lane half), col = lane&15
  #pragma unroll
  for (int mi = 0; mi < 4; ++mi)
    #pragma unroll
    for (int ni = 0; ni < 2; ++ni)
      #pragma unroll
      for (int i = 0; i < 8; ++i) {
        const int r  = bm + wm + mi * 16 + sel * 8 + i;
        const int cc = bn + wn + ni * 16 + lrow;
        const float vv = acc[mi][ni][i];
        if constexpr (OUT_F32) {
          ((float*)Cout)[(size_t)r * N + cc] = vv + (bias ? bias[cc] : 0.f);
        } else {
          ((_Float16*)Cout)[(size_t)r * N + cc] = (_Float16)vv;
        }
      }
}

// ---------------------------------------------------------------------------
// RoPE (interleaved pairs) + rearrange (b,t,h,hd) -> (b,h,t,hd), f16
// ---------------------------------------------------------------------------
__global__ __launch_bounds__(256)
void rope_kernel(const _Float16* __restrict__ src, const float* __restrict__ cosT,
                 const float* __restrict__ sinT, _Float16* __restrict__ dst,
                 int heads)
{
  const int rowW = heads * HD_;
  int idx = blockIdx.x * blockDim.x + threadIdx.x;
  const int p = idx & 63;
  int r = idx >> 6;
  const int t  = r % T_;  r /= T_;
  const int hh = r % heads;
  const int b  = r / heads;
  const size_t si = (size_t)(b * T_ + t) * rowW + hh * HD_ + 2 * p;
  const float x1 = (float)src[si], x2 = (float)src[si + 1];
  const float c = cosT[t * 64 + p], s = sinT[t * 64 + p];
  const size_t di = ((size_t)(b * heads + hh) * T_ + t) * HD_ + 2 * p;
  dst[di]     = (_Float16)(x1 * c - x2 * s);
  dst[di + 1] = (_Float16)(x1 * s + x2 * c);
}

// V: (b,t,kv,hd) -> (b,kv,hd,t) so PV B-fragments are contiguous along t
__global__ __launch_bounds__(256)
void vtrans_kernel(const _Float16* __restrict__ src, _Float16* __restrict__ dst)
{
  int idx = blockIdx.x * blockDim.x + threadIdx.x;   // (b,kv,d,t) flat
  const int t = idx & (T_ - 1);
  int r = idx >> 11;
  const int d  = r & 127; r >>= 7;
  const int kv = r & 3;
  const int b  = r >> 2;
  dst[idx] = src[(size_t)(b * T_ + t) * (KVH_ * HD_) + kv * HD_ + d];
}

// ---------------------------------------------------------------------------
// Flash attention (causal, GQA): one wave per 16-row Q tile; the 8 waves of a
// block share one (b, h) and cooperatively stage K/V tiles into LDS with
// double-buffered global_load_async_to_lds_b128 DMAs (ASYNCcnt).
// q: (b,H,T,HD) f16 | k: (b,KVH,T,HD) f16 | v: (b,KVH,HD,T) f16
// ctx out: (b,T,H*HD) f16
// ---------------------------------------------------------------------------
#define KSTR 144   // K tile LDS row stride (halves): 288B, 16B-multiple
#define VSTR 48    // V tile LDS row stride (halves):  96B, 16B-multiple

__global__ __launch_bounds__(256)
void attn_kernel(const _Float16* __restrict__ q, const _Float16* __restrict__ k,
                 const _Float16* __restrict__ v, _Float16* __restrict__ ctx)
{
  __shared__ __align__(16) _Float16 Kbuf[2][32 * KSTR];    // 2 x 9 KB
  __shared__ __align__(16) _Float16 Vbuf[2][128 * VSTR];   // 2 x 12 KB
  __shared__ __align__(16) _Float16 pbuf_all[8][16 * 32];  // 8 KB

  const int tid  = threadIdx.x;
  const int w    = tid >> 5, lane = tid & 31;
  const int lrow = lane & 15, sel = lane >> 4;
  const int b    = blockIdx.z, h = blockIdx.y;
  const int q0   = blockIdx.x * 128 + w * 16;
  const int kv   = h >> 2;                       // G = H/KVH = 4
  _Float16* pbuf = pbuf_all[w];

  const _Float16* qb = q + ((size_t)(b * H_ + h) * T_ + q0) * HD_;
  const _Float16* kb = k + (size_t)(b * KVH_ + kv) * T_ * HD_;
  const _Float16* vb = v + (size_t)(b * KVH_ + kv) * HD_ * T_;

  // Q fragments (A layout: M = lane&15, K halves = sel*16 + {0..15})
  v16h aq[4];
  #pragma unroll
  for (int kc = 0; kc < 4; ++kc) {
    const _Float16* base = qb + (size_t)lrow * HD_ + kc * 32;
    aq[kc] = cat8(*(const v8h*)(base + sel * 16),
                  *(const v8h*)(base + sel * 16 + 8));
  }

  const v8f zf = {0.f,0.f,0.f,0.f,0.f,0.f,0.f,0.f};
  v8f o[8];
  float m_i[8], l_i[8];
  #pragma unroll
  for (int i = 0; i < 8; ++i) { o[i] = zf; m_i[i] = -1e30f; l_i[i] = 0.f; }

  const float scale = 0.08838834764831845f;      // 1/sqrt(128)
  const int nblk_w = (q0 + 15) / 32 + 1;                     // this wave
  const int nblk_b = ((int)blockIdx.x * 128 + 127) / 32 + 1; // whole block

  // ---- stage issuer: 4 async b128 DMAs per wave (2 for K, 2 for V) ----
  auto issue_kv = [&](int j0, int buf) {
    _Float16* Ks = Kbuf[buf];
    _Float16* Vs = Vbuf[buf];
    #pragma unroll
    for (int c = 0; c < 2; ++c) {           // K: 32 rows x 128 halves
      int idx = tid + c * 256;              // 512 chunks of 8 halves
      int row = idx >> 4, c8 = (idx & 15) << 3;
      async_ld16(kb + (size_t)(j0 + row) * HD_ + c8,
                 (unsigned)(uintptr_t)(Ks + row * KSTR + c8));
    }
    #pragma unroll
    for (int c = 0; c < 2; ++c) {           // V: 128 rows x 32 halves
      int idx = tid + c * 256;              // 512 chunks of 8 halves
      int d = idx >> 2, c8 = (idx & 3) << 3;
      async_ld16(vb + (size_t)d * T_ + j0 + c8,
                 (unsigned)(uintptr_t)(Vs + d * VSTR + c8));
    }
  };

  issue_kv(0, 0);                            // prologue: stage block 0

  for (int jb = 0; jb < nblk_b; ++jb) {
    const int j0 = jb * 32;
    if (jb + 1 < nblk_b) { issue_kv(j0 + 32, (jb + 1) & 1); wait_async(1); }
    else                 { wait_async(0); }
    __syncthreads();                          // tile jb visible to all waves

    if (jb < nblk_w) {                        // wave-uniform causal predicate
      const _Float16* Ks = Kbuf[jb & 1];
      const _Float16* Vs = Vbuf[jb & 1];

      // ---- logits = Q @ K^T, 2 N-tiles x 4 K-steps ----
      v8f c0 = zf, c1 = zf;
      #pragma unroll
      for (int kc = 0; kc < 4; ++kc) {
        const _Float16* r0 = Ks + (size_t)lrow * KSTR + kc * 32 + sel * 16;
        c0 = wmma16(aq[kc], load16(r0), c0);
        const _Float16* r1 = Ks + (size_t)(16 + lrow) * KSTR + kc * 32 + sel * 16;
        c1 = wmma16(aq[kc], load16(r1), c1);
      }

      // ---- causal mask + online softmax (C layout, 16-lane row reductions) ----
      float p0a[8], p1a[8], corr[8];
      const int colg0 = j0 + lrow, colg1 = j0 + 16 + lrow;
      #pragma unroll
      for (int i = 0; i < 8; ++i) {
        const int rowg = q0 + sel * 8 + i;
        const float x0 = (colg0 <= rowg) ? c0[i] * scale : -1e30f;
        const float x1 = (colg1 <= rowg) ? c1[i] * scale : -1e30f;
        float mx = fmaxf(x0, x1);
        #pragma unroll
        for (int d = 1; d < 16; d <<= 1) mx = fmaxf(mx, __shfl_xor(mx, d, 32));
        const float nm = fmaxf(m_i[i], mx);
        corr[i] = __expf(m_i[i] - nm);
        const float p0 = __expf(x0 - nm), p1 = __expf(x1 - nm);
        float rs = p0 + p1;
        #pragma unroll
        for (int d = 1; d < 16; d <<= 1) rs += __shfl_xor(rs, d, 32);
        l_i[i] = l_i[i] * corr[i] + rs;
        m_i[i] = nm;
        p0a[i] = p0; p1a[i] = p1;
      }
      #pragma unroll
      for (int ht = 0; ht < 8; ++ht)
        #pragma unroll
        for (int i = 0; i < 8; ++i) o[ht][i] *= corr[i];

      // ---- reshape P: C layout -> A fragment via per-wave LDS tile ----
      __builtin_amdgcn_wave_barrier();
      #pragma unroll
      for (int i = 0; i < 8; ++i) {
        const int r = sel * 8 + i;
        pbuf[r * 32 + lrow]      = (_Float16)p0a[i];
        pbuf[r * 32 + 16 + lrow] = (_Float16)p1a[i];
      }
      __builtin_amdgcn_wave_barrier();
      asm volatile("s_wait_dscnt 0x0" ::: "memory");

      const _Float16* pb = &pbuf[lrow * 32 + sel * 16];
      const v16h pa = cat8(*(const v8h*)pb, *(const v8h*)(pb + 8));

      // ---- O += P @ V (V staged (hd, t): contiguous B fragments) ----
      #pragma unroll
      for (int ht = 0; ht < 8; ++ht) {
        const _Float16* vr = Vs + (size_t)(ht * 16 + lrow) * VSTR + sel * 16;
        o[ht] = wmma16(pa, load16(vr), o[ht]);
      }
    }
    __syncthreads();                          // all reads done before reuse
  }

  // ---- normalize and write ctx (b, t, h*HD) f16 ----
  float inv[8];
  #pragma unroll
  for (int i = 0; i < 8; ++i) inv[i] = 1.f / l_i[i];
  #pragma unroll
  for (int ht = 0; ht < 8; ++ht)
    #pragma unroll
    for (int i = 0; i < 8; ++i) {
      const int rowg = q0 + sel * 8 + i;
      const int colg = h * HD_ + ht * 16 + lrow;
      ctx[(size_t)(b * T_ + rowg) * DOUT + colg] = (_Float16)(o[ht][i] * inv[i]);
    }
}

// ---------------------------------------------------------------------------
extern "C" void kernel_launch(void* const* d_in, const int* in_sizes, int n_in,
                              void* d_out, int out_size, void* d_ws, size_t ws_size,
                              hipStream_t stream)
{
  (void)in_sizes; (void)n_in; (void)out_size; (void)ws_size;
  const float* x    = (const float*)d_in[0];
  const float* Wq   = (const float*)d_in[1];
  const float* Wk   = (const float*)d_in[2];
  const float* Wv   = (const float*)d_in[3];
  const float* Wo   = (const float*)d_in[4];
  const float* bo   = (const float*)d_in[5];
  const float* cosT = (const float*)d_in[6];
  const float* sinT = (const float*)d_in[7];
  float* out = (float*)d_out;

  const size_t M1 = (size_t)B_ * T_;              // 4096 rows
  _Float16* q_raw = (_Float16*)d_ws;              //  8M halves
  _Float16* k_raw = q_raw + M1 * DOUT;            // +2M
  _Float16* v_raw = k_raw + M1 * (KVH_ * HD_);    // +2M
  _Float16* q_ws  = v_raw + M1 * (KVH_ * HD_);    // +8M
  _Float16* k_ws  = q_ws  + M1 * DOUT;            // +2M
  _Float16* v_ws  = k_ws  + M1 * (KVH_ * HD_);    // +2M
  _Float16* ctx   = v_ws  + M1 * (KVH_ * HD_);    // +8M  (total 32M halves = 64MB)

  const dim3 blk(256);

  // Q/K/V projections (fp32 in -> f16 out, f32 accumulate in WMMA)
  gemm_wmma<false, false><<<dim3(DOUT / 128, M1 / 128), blk, 0, stream>>>(
      x, Wq, q_raw, nullptr, (int)M1, DOUT, DIN);
  gemm_wmma<false, false><<<dim3((KVH_ * HD_) / 128, M1 / 128), blk, 0, stream>>>(
      x, Wk, k_raw, nullptr, (int)M1, KVH_ * HD_, DIN);
  gemm_wmma<false, false><<<dim3((KVH_ * HD_) / 128, M1 / 128), blk, 0, stream>>>(
      x, Wv, v_raw, nullptr, (int)M1, KVH_ * HD_, DIN);

  // RoPE + layout rearrange
  rope_kernel<<<(B_ * H_ * T_ * 64) / 256, blk, 0, stream>>>(q_raw, cosT, sinT, q_ws, H_);
  rope_kernel<<<(B_ * KVH_ * T_ * 64) / 256, blk, 0, stream>>>(k_raw, cosT, sinT, k_ws, KVH_);
  vtrans_kernel<<<(B_ * KVH_ * HD_ * T_) / 256, blk, 0, stream>>>(v_raw, v_ws);

  // Causal GQA flash attention (async-DMA staged K/V)
  attn_kernel<<<dim3(T_ / 128, H_, B_), blk, 0, stream>>>(q_ws, k_ws, v_ws, ctx);

  // Output projection + bias (f16 A, fp32 out)
  gemm_wmma<true, true><<<dim3(DOUT / 128, M1 / 128), blk, 0, stream>>>(
      ctx, Wo, out, bo, (int)M1, DOUT, DOUT);
}
